// WindowAttention_14748917694823
// MI455X (gfx1250) — matrix operand order repaired and checked by reading, not compiled, not run on hardware
//
#include <hip/hip_runtime.h>
#include <hip/hip_bf16.h>
#include <math.h>

typedef __attribute__((ext_vector_type(16))) __bf16 bf16x16;
typedef __attribute__((ext_vector_type(8)))  float  f32x8;

#define DIM 256
#define H   8
#define HD  32
#define W   64
#define NW  128
#define TX  40          // 2*L_XYZ
#define TR  32          // 2*L_RGB
#define NTX (3*TX)      // 120
#define PX  128         // NTX padded to a multiple of 32 (K-step) / 16 (N-tile)
#define NTR (3*TR)      // 96

__device__ __forceinline__ f32x8 wmma_bf16(bf16x16 a, bf16x16 b, f32x8 c) {
  // 8 args: (neg_a, A, neg_b, B, c_mod, C, reuse_a, reuse_b)
  return __builtin_amdgcn_wmma_f32_16x16x32_bf16(false, a, false, b, (short)0, c, false, false);
}

// ---- WMMA operand builders (16-bit A 16x32 / B 32x16 lane layouts in contiguous k = 16*half + s order, same for A and B)
// A: lanes0-15 M=lane K in {0..15}; lanes16-31 K in {16..31}
__device__ __forceinline__ bf16x16 a_from_lds_bf16(const __bf16* s, int stride,
                                                   int l15, int half, int kofs) {
  bf16x16 a;
  #pragma unroll
  for (int v = 0; v < 8; ++v) {
    const int k0 = kofs + 2*v + 16*half;
    a[2*v]   = s[l15*stride + k0];
    a[2*v+1] = s[l15*stride + k0 + 1];
  }
  return a;
}
__device__ __forceinline__ bf16x16 a_from_lds_f32(const float* s, int stride,
                                                  int l15, int half, int kofs) {
  bf16x16 a;
  #pragma unroll
  for (int v = 0; v < 8; ++v) {
    const int k0 = kofs + 2*v + 16*half;
    a[2*v]   = (__bf16)s[l15*stride + k0];
    a[2*v+1] = (__bf16)s[l15*stride + k0 + 1];
  }
  return a;
}
// B: lanes0-15 N=lane K in {0..15}; lanes16-31 K in {16..31}; pairs per VGPR.
// Column-row variant: this lane's column is contiguous over k at `row`.
__device__ __forceinline__ bf16x16 b_from_row_f32(const float* row, int half, int kofs) {
  bf16x16 b;
  #pragma unroll
  for (int v = 0; v < 8; ++v) {
    const int k0 = kofs + 2*v + 16*half;
    const float2 f = *(const float2*)(row + k0);
    b[2*v] = (__bf16)f.x;  b[2*v+1] = (__bf16)f.y;
  }
  return b;
}
__device__ __forceinline__ bf16x16 b_from_row_bf16(const __bf16* row, int half, int kofs) {
  bf16x16 b;
  #pragma unroll
  for (int v = 0; v < 8; ++v) {
    const int k0 = kofs + 2*v + 16*half;
    b[2*v] = row[k0];  b[2*v+1] = row[k0 + 1];
  }
  return b;
}
// K-major variant: B[k][n] = base[k*kstride + n] for this lane's fixed n.
__device__ __forceinline__ bf16x16 b_from_kmajor_f32(const float* base, int kstride,
                                                     int n, int half, int kofs) {
  bf16x16 b;
  #pragma unroll
  for (int v = 0; v < 8; ++v) {
    const int k0 = kofs + 2*v + 16*half;
    b[2*v]   = (__bf16)base[(size_t)k0 * kstride + n];
    b[2*v+1] = (__bf16)base[(size_t)(k0+1) * kstride + n];
  }
  return b;
}
__device__ __forceinline__ bf16x16 b_from_kmajor_bf16(const __bf16* base, int kstride,
                                                      int n, int half, int kofs) {
  bf16x16 b;
  #pragma unroll
  for (int v = 0; v < 8; ++v) {
    const int k0 = kofs + 2*v + 16*half;
    b[2*v]   = base[k0 * kstride + n];
    b[2*v+1] = base[(k0+1) * kstride + n];
  }
  return b;
}

// ---------------------------------------------------------------------------
// out[m][n] = (sum_k A[m][k]*Wt[n][k] + bias[n]) * (n<scale_cols ? scale : 1)
// ---------------------------------------------------------------------------
__global__ __launch_bounds__(256) void gemm_bias_wmma(
    const float* __restrict__ A, const float* __restrict__ Wt,
    const float* __restrict__ bias, float* __restrict__ out,
    int M, int N, int K, int scale_cols, float scale)
{
  const int lane = threadIdx.x & 31;
  const int wave = threadIdx.x >> 5;
  const int half = lane >> 4;
  const int l15  = lane & 15;

  const int m0 = blockIdx.y * 64 + (wave >> 1) * 16;
  const int n0 = blockIdx.x * 64 + (wave & 1) * 32;

  f32x8 c0 = {}; f32x8 c1 = {};

  const float* Arow = A + (size_t)(m0 + l15) * K;
  const float* W0   = Wt + (size_t)(n0 + l15) * K;
  const float* W1   = Wt + (size_t)(n0 + 16 + l15) * K;

  for (int kb = 0; kb < K; kb += 32) {
    bf16x16 av;
    #pragma unroll
    for (int v = 0; v < 8; ++v) {
      const int k0 = kb + 2*v + 16*half;
      const float2 f = *(const float2*)(Arow + k0);
      av[2*v] = (__bf16)f.x;  av[2*v+1] = (__bf16)f.y;
    }
    const bf16x16 b0v = b_from_row_f32(W0, half, kb);
    const bf16x16 b1v = b_from_row_f32(W1, half, kb);
    if (kb + 32 < K) __builtin_prefetch(Arow + kb + 32, 0, 1); // global_prefetch_b8
    c0 = wmma_bf16(av, b0v, c0);
    c1 = wmma_bf16(av, b1v, c1);
  }

  #pragma unroll
  for (int v = 0; v < 8; ++v) {
    const int m  = m0 + v + 8*half;
    const int n  = n0 + l15;
    const int n1 = n + 16;
    const float s0 = (n  < scale_cols) ? scale : 1.0f;
    const float s1 = (n1 < scale_cols) ? scale : 1.0f;
    out[(size_t)m * N + n ] = (c0[v] + bias[n ]) * s0;
    out[(size_t)m * N + n1] = (c1[v] + bias[n1]) * s1;
  }
}

// ---------------------------------------------------------------------------
// One workgroup per (window, head).  256 threads = 8 waves.  All four
// GEMM-shaped einsums (QK^T, dq/dk bias tables, attn@V, value-bias) run on
// the WMMA pipe; only index math / softmax / scatter stay on VALU.
// ---------------------------------------------------------------------------
__global__ __launch_bounds__(256) void window_attn_kernel(
    const float* __restrict__ qkv,      // [8192][768], q already scaled
    const float* __restrict__ coords,   // [8192][6]
    const float* __restrict__ qxt, const float* __restrict__ kxt,
    const float* __restrict__ vxt,      // [3][40][8][32]
    const float* __restrict__ qrt, const float* __restrict__ krt,
    const float* __restrict__ vrt,      // [3][32][8][32]
    float* __restrict__ outp)           // [8192][256]
{
  extern __shared__ char smem[];
  __bf16* qb = (__bf16*)smem;               // [64][32]
  __bf16* kb = qb + W*HD;
  __bf16* vb = kb + W*HD;
  float* sc  = (float*)(vb + W*HD);         // [64][6]
  float* dqx = sc + W*6;                    // [64][128] (cols 120..127 pad; reused as at_x)
  float* dkx = dqx + W*PX;                  // [64][128]
  float* dqr = dkx + W*PX;                  // [64][96]  (reused as at_r)
  float* dkr = dqr + W*NTR;                 // [64][96]
  float* att = dkr + W*NTR;                 // [64][64]

  const int head = blockIdx.x;
  const int win  = blockIdx.y;
  const int tid  = threadIdx.x;
  const int lane = tid & 31;
  const int wave = tid >> 5;
  const int half = lane >> 4;
  const int l15  = lane & 15;

  // ---- stage q/k/v (bf16) and quantize-scaled coords into LDS
  const size_t base = (size_t)win * W * 768 + head * HD;
  for (int p = tid; p < W*HD; p += 256) {
    const int i = p >> 5, d = p & 31;
    const float* row = qkv + base + (size_t)i * 768 + d;
    qb[p] = (__bf16)row[0];
    kb[p] = (__bf16)row[256];
    vb[p] = (__bf16)row[512];
  }
  for (int p = tid; p < W*6; p += 256) {
    const int c = p % 6;
    sc[p] = coords[(size_t)win*W*6 + p] * (c < 3 ? 4.0f : 8.0f);  // QS / CQS
  }
  __syncthreads();

  // ---- logits = q*k^T : 16 tiles of 16x16x32, 2 per wave
  #pragma unroll
  for (int tt = 0; tt < 2; ++tt) {
    const int tile = wave*2 + tt;
    const int ti = tile >> 2, tj = tile & 3;
    const bf16x16 av = a_from_lds_bf16(qb + ti*16*HD, HD, l15, half, 0);
    const bf16x16 bv = b_from_row_bf16(kb + (tj*16 + l15)*HD, half, 0); // B[d][j]=k[j][d]
    f32x8 c = {};
    c = wmma_bf16(av, bv, c);
    #pragma unroll
    for (int v = 0; v < 8; ++v)
      att[(ti*16 + v + 8*half)*W + tj*16 + l15] = c[v];
  }

  // ---- dq/dk bias tables on WMMA: [64x32] @ [32 x NTX|NTR]
  {
    // xyz: 8 N-tiles (N padded 120->128); wave = n-tile
    const int ct  = wave*16 + l15;
    const int ctc = (ct < NTX) ? ct : (NTX-1);
    const bf16x16 bq = b_from_row_f32(qxt + (ctc*H + head)*HD, half, 0);
    const bf16x16 bk = b_from_row_f32(kxt + (ctc*H + head)*HD, half, 0);
    #pragma unroll
    for (int it = 0; it < 4; ++it) {
      const bf16x16 aq = a_from_lds_bf16(qb + it*16*HD, HD, l15, half, 0);
      f32x8 cq = {}; cq = wmma_bf16(aq, bq, cq);
      const bf16x16 ak = a_from_lds_bf16(kb + it*16*HD, HD, l15, half, 0);
      f32x8 ck = {}; ck = wmma_bf16(ak, bk, ck);
      #pragma unroll
      for (int v = 0; v < 8; ++v) {
        const int i = it*16 + v + 8*half;
        if (ct < NTX) { dqx[i*PX + ct] = cq[v]; dkx[i*PX + ct] = ck[v]; }
      }
    }
  }
  if (wave < 6) {   // rgb: 6 N-tiles exactly
    const int ct = wave*16 + l15;
    const bf16x16 bq = b_from_row_f32(qrt + (ct*H + head)*HD, half, 0);
    const bf16x16 bk = b_from_row_f32(krt + (ct*H + head)*HD, half, 0);
    #pragma unroll
    for (int it = 0; it < 4; ++it) {
      const bf16x16 aq = a_from_lds_bf16(qb + it*16*HD, HD, l15, half, 0);
      f32x8 cq = {}; cq = wmma_bf16(aq, bq, cq);
      const bf16x16 ak = a_from_lds_bf16(kb + it*16*HD, HD, l15, half, 0);
      f32x8 ck = {}; ck = wmma_bf16(ak, bk, ck);
      #pragma unroll
      for (int v = 0; v < 8; ++v) {
        const int i = it*16 + v + 8*half;
        dqr[i*NTR + ct] = cq[v];  dkr[i*NTR + ct] = ck[v];
      }
    }
  }
  __syncthreads();

  // ---- add relative-position bias to logits (gather from dq/dk)
  for (int p = tid; p < W*W; p += 256) {
    const int i = p >> 6, j = p & 63;
    float b = 0.0f;
    #pragma unroll
    for (int c = 0; c < 3; ++c) {
      int ix = (int)floorf(sc[i*6+c] - sc[j*6+c]) + 20;
      ix = ix < 0 ? 0 : (ix > 39 ? 39 : ix);
      b += dqx[i*PX + c*TX + ix] + dkx[j*PX + c*TX + ix];
      int ir = (int)floorf(sc[i*6+3+c] - sc[j*6+3+c]) + 16;
      ir = ir < 0 ? 0 : (ir > 31 ? 31 : ir);
      b += dqr[i*NTR + c*TR + ir] + dkr[j*NTR + c*TR + ir];
    }
    att[p] += b;
  }
  __syncthreads();

  // ---- dq buffers dead: reuse as scatter accumulators (zero incl. pad cols)
  for (int p = tid; p < W*PX;  p += 256) dqx[p] = 0.0f;
  for (int p = tid; p < W*NTR; p += 256) dqr[p] = 0.0f;
  __syncthreads();

  // ---- softmax per row + scatter attn into (c,t) bins (row-private)
  if (tid < W) {
    const int i = tid;
    float m = -3.4e38f;
    for (int j = 0; j < W; ++j) m = fmaxf(m, att[i*W+j]);
    float s = 0.0f;
    for (int j = 0; j < W; ++j) { float e = __expf(att[i*W+j] - m); att[i*W+j] = e; s += e; }
    const float inv = 1.0f / s;
    for (int j = 0; j < W; ++j) {
      const float a = att[i*W+j] * inv;
      att[i*W+j] = a;
      #pragma unroll
      for (int c = 0; c < 3; ++c) {
        int ix = (int)floorf(sc[i*6+c] - sc[j*6+c]) + 20;
        ix = ix < 0 ? 0 : (ix > 39 ? 39 : ix);
        dqx[i*PX + c*TX + ix] += a;
        int ir = (int)floorf(sc[i*6+3+c] - sc[j*6+3+c]) + 16;
        ir = ir < 0 ? 0 : (ir > 31 ? 31 : ir);
        dqr[i*NTR + c*TR + ir] += a;
      }
    }
  }
  __syncthreads();

  // ---- out = attn@v + at_x@Vx + at_r@Vr, all WMMA into one C fragment
  {
    const int ti = wave >> 1;   // i-tile 0..3
    const int td = wave & 1;    // d-tile 0..1
    const int d  = td*16 + l15;
    f32x8 c = {};
    #pragma unroll
    for (int ks = 0; ks < 2; ++ks) {          // attn @ v : K=64
      const bf16x16 av = a_from_lds_f32(att + ti*16*W, W, l15, half, ks*32);
      const bf16x16 bv = b_from_kmajor_bf16(vb, HD, d, half, ks*32);
      c = wmma_bf16(av, bv, c);
    }
    #pragma unroll
    for (int ks = 0; ks < 4; ++ks) {          // at_x @ Vx : K=128 (zero-padded)
      const bf16x16 av = a_from_lds_f32(dqx + ti*16*PX, PX, l15, half, ks*32);
      const bf16x16 bv = b_from_kmajor_f32(vxt + head*HD, H*HD, d, half, ks*32);
      c = wmma_bf16(av, bv, c);
    }
    #pragma unroll
    for (int ks = 0; ks < 3; ++ks) {          // at_r @ Vr : K=96
      const bf16x16 av = a_from_lds_f32(dqr + ti*16*NTR, NTR, l15, half, ks*32);
      const bf16x16 bv = b_from_kmajor_f32(vrt + head*HD, H*HD, d, half, ks*32);
      c = wmma_bf16(av, bv, c);
    }
    #pragma unroll
    for (int v = 0; v < 8; ++v) {
      const int i = ti*16 + v + 8*half;
      outp[((size_t)win*W + i)*DIM + head*HD + d] = c[v];
    }
  }
}

extern "C" void kernel_launch(void* const* d_in, const int* in_sizes, int n_in,
                              void* d_out, int out_size, void* d_ws, size_t ws_size,
                              hipStream_t stream) {
  (void)in_sizes; (void)n_in; (void)out_size; (void)ws_size;
  const float* feats  = (const float*)d_in[0];
  const float* coords = (const float*)d_in[1];
  const float* qkv_w  = (const float*)d_in[2];
  const float* qkv_b  = (const float*)d_in[3];
  const float* qxt    = (const float*)d_in[4];
  const float* kxt    = (const float*)d_in[5];
  const float* vxt    = (const float*)d_in[6];
  const float* qrt    = (const float*)d_in[7];
  const float* krt    = (const float*)d_in[8];
  const float* vrt    = (const float*)d_in[9];
  const float* proj_w = (const float*)d_in[10];
  const float* proj_b = (const float*)d_in[11];
  float* out = (float*)d_out;

  float* ws_qkv = (float*)d_ws;                        // [8192][768] fp32
  float* ws_att = ws_qkv + (size_t)8192 * 768;         // [8192][256] fp32

  const float scale = 0.17677669529663687f;            // HD^-0.5

  gemm_bias_wmma<<<dim3(768/64, 8192/64), 256, 0, stream>>>(
      feats, qkv_w, qkv_b, ws_qkv, 8192, 768, 256, 256, scale);

  const size_t smem = (size_t)(3*W*HD*2) + W*6*4
                    + (size_t)W*(2*PX + 2*NTR)*4 + (size_t)W*W*4;  // 144896 B
  window_attn_kernel<<<dim3(H, NW), 256, smem, stream>>>(
      ws_qkv, coords, qxt, kxt, vxt, qrt, krt, vrt, ws_att);

  gemm_bias_wmma<<<dim3(256/64, 8192/64), 256, 0, stream>>>(
      ws_att, proj_w, proj_b, out, 8192, 256, 256, 0, 1.0f);
}
